// DetectionDecoderLayer_53747220742150
// MI455X (gfx1250) — matrix operand order repaired and checked by reading, not compiled, not run on hardware
//
#include <hip/hip_runtime.h>
#include <hip/hip_bf16.h>

typedef __attribute__((ext_vector_type(16))) _Float16 v16h;
typedef __attribute__((ext_vector_type(8)))  float    v8f;

#define WMMA16(a, b, c) __builtin_amdgcn_wmma_f32_16x16x32_f16(false, (a), false, (b), (short)0, (c), false, false)

// ---------------- constants ----------------
constexpr int Bn   = 8;
constexpr int NQn  = 900;
constexpr int Dn   = 256;
constexpr int Hn   = 8;
constexpr int Pn   = 4;
constexpr int BEVn = 200;
constexpr int FFNn = 512;
constexpr int HDn  = 32;        // D / H
constexpr int Mn   = Bn * NQn;  // 7200 rows, divisible by 16
constexpr int SPn  = 928;       // padded sequence (29*32, >=900)
constexpr int QTn  = 57;        // ceil(900/16)

// ---------------- LayerNorm (optionally with additive input) ----------------
__global__ void ln_kernel(const float* __restrict__ x, const float* __restrict__ addsrc,
                          const float* __restrict__ g, const float* __restrict__ bt,
                          float* __restrict__ out) {
  __shared__ float red[256];
  int row = blockIdx.x, t = threadIdx.x;
  size_t base = (size_t)row * Dn;
  float v = x[base + t];
  if (addsrc) v += addsrc[base + t];
  red[t] = v; __syncthreads();
  for (int s = 128; s > 0; s >>= 1) { if (t < s) red[t] += red[t + s]; __syncthreads(); }
  float mean = red[0] / (float)Dn;
  __syncthreads();
  float d = v - mean;
  red[t] = d * d; __syncthreads();
  for (int s = 128; s > 0; s >>= 1) { if (t < s) red[t] += red[t + s]; __syncthreads(); }
  float var = red[0] / (float)Dn;
  out[base + t] = d * rsqrtf(var + 1e-5f) * g[t] + bt[t];
}

// ---------------- pack f32 [M,K] row-major -> WMMA A-fragment layout (f16) ----------------
__global__ void pack_a_kernel(const float* __restrict__ src, _Float16* __restrict__ dst,
                              int M, int K, int total) {
  int idx = blockIdx.x * blockDim.x + threadIdx.x;
  if (idx >= total) return;
  int KT = K >> 5;
  int i = idx & 15;
  int lane = (idx >> 4) & 31;
  int t = idx >> 9;
  int kt = t % KT, mt = t / KT;
  int mrow = mt * 16 + (lane & 15);
  int hi = lane >> 4;
  int v = i >> 1;
  int kk = kt * 32 + hi * 16 + i;
  dst[idx] = (_Float16)src[(size_t)mrow * K + kk];
}

// ---------------- pack weight W[N,K] row-major -> WMMA B-fragment layout (f16) ----------------
__global__ void pack_b_kernel(const float* __restrict__ w, _Float16* __restrict__ dst,
                              int N, int K, int total) {
  int idx = blockIdx.x * blockDim.x + threadIdx.x;
  if (idx >= total) return;
  int KT = K >> 5;
  int i = idx & 15;
  int lane = (idx >> 4) & 31;
  int t = idx >> 9;
  int kt = t % KT, nt = t / KT;
  int ncol = nt * 16 + (lane & 15);
  int hi = lane >> 4;
  int kk = kt * 32 + hi * 16 + i;
  dst[idx] = (_Float16)w[(size_t)ncol * K + kk];
}

// ---------------- single-wave WMMA GEMM (used when N % 64 != 0) ----------------
// act: 0 none, 1 relu, 2 tanh*0.2
__global__ void gemm_wmma_kernel(const _Float16* __restrict__ Ap, const _Float16* __restrict__ Bp,
                                 const float* __restrict__ bias, const float* __restrict__ res,
                                 float* __restrict__ out, int M, int N, int K, int act) {
  int lane = threadIdx.x;           // 32 threads = 1 wave
  int nt = blockIdx.x, mt = blockIdx.y;
  int KT = K >> 5;
  v8f acc = {};
  const v16h* Abase = (const v16h*)Ap + (size_t)(mt * KT) * 32 + lane;
  const v16h* Bbase = (const v16h*)Bp + (size_t)(nt * KT) * 32 + lane;
  for (int kt = 0; kt < KT; ++kt) {
    v16h a = Abase[(size_t)kt * 32];
    v16h b = Bbase[(size_t)kt * 32];
    acc = WMMA16(a, b, acc);
  }
  int n = nt * 16 + (lane & 15);
  int rbase = mt * 16 + (lane >> 4) * 8;
  float bv = bias ? bias[n] : 0.f;
  for (int r = 0; r < 8; ++r) {
    int m = rbase + r;
    float v = acc[r] + bv;
    if (act == 1) v = fmaxf(v, 0.f);
    else if (act == 2) v = tanhf(v) * 0.2f;
    if (res) v += res[(size_t)m * N + n];
    out[(size_t)m * N + n] = v;
  }
}

// ---------------- 4-wave WMMA GEMM with async-to-LDS staging of the shared A fragment ------
// Block = 16M x 64N. Wave w owns n-tile (blockIdx.x*4 + w). Wave 0 stages the A fragment
// chunk (1KB per K-step) into LDS with global_load_async_to_lds_b128 (ASYNCcnt), then all
// waves read it back with ds_read_b128 while streaming their own B fragments from global.
__global__ void gemm_wmma4_kernel(const _Float16* __restrict__ Ap, const _Float16* __restrict__ Bp,
                                  const float* __restrict__ bias, const float* __restrict__ res,
                                  float* __restrict__ out, int M, int N, int K, int act) {
  __shared__ _Float16 sA[512];     // 32 lanes * 16 halves = 1KB
  int tid  = threadIdx.x;
  int wave = tid >> 5;
  int lane = tid & 31;
  int nt = blockIdx.x * 4 + wave;
  int mt = blockIdx.y;
  int KT = K >> 5;
  v8f acc = {};
  const _Float16* Achunk0 = Ap + (size_t)(mt * KT) * 512;   // + kt*512 + lane*16
  const v16h* Bbase = (const v16h*)Bp + (size_t)(nt * KT) * 32 + lane;
  unsigned lds_off = (unsigned)(size_t)(&sA[lane * 16]);    // flat LDS aperture: low 32 bits = LDS offset

  for (int kt = 0; kt < KT; ++kt) {
    __syncthreads();               // protect LDS from previous iteration's readers
    if (wave == 0) {
      unsigned long long ga = (unsigned long long)(size_t)(Achunk0 + (size_t)kt * 512 + lane * 16);
      asm volatile("global_load_async_to_lds_b128 %0, %1, off"
                   :: "v"(lds_off), "v"(ga) : "memory");
      asm volatile("global_load_async_to_lds_b128 %0, %1, off offset:16"
                   :: "v"(lds_off), "v"(ga) : "memory");
      asm volatile("s_wait_asynccnt 0x0" ::: "memory");
    }
    __syncthreads();
    v16h a = *(const v16h*)(&sA[lane * 16]);
    v16h b = Bbase[(size_t)kt * 32];
    acc = WMMA16(a, b, acc);
  }

  int n = nt * 16 + (lane & 15);
  int rbase = mt * 16 + (lane >> 4) * 8;
  float bv = bias ? bias[n] : 0.f;
  for (int r = 0; r < 8; ++r) {
    int m = rbase + r;
    float v = acc[r] + bv;
    if (act == 1) v = fmaxf(v, 0.f);
    else if (act == 2) v = tanhf(v) * 0.2f;
    if (res) v += res[(size_t)m * N + n];
    out[(size_t)m * N + n] = v;
  }
}

// ---------------- qkv split: f32 [B,NQ,3D] -> Qf16 (scaled), Kf16 [B,H,SP,HD], Vt [B,H,HD,SP] ----------------
__global__ void qkv_split_kernel(const float* __restrict__ qkv, _Float16* __restrict__ Qf,
                                 _Float16* __restrict__ Kf, _Float16* __restrict__ Vt) {
  int idx = blockIdx.x * blockDim.x + threadIdx.x;
  int total = Bn * Hn * SPn * HDn;
  if (idx >= total) return;
  int d  = idx & 31;
  int s  = (idx >> 5) % SPn;
  int bh = idx / (SPn * HDn);
  int b = bh >> 3, h = bh & 7;
  const float scale = 0.17677669529663687f;  // 1/sqrt(32)
  float q = 0.f, k = 0.f, v = 0.f;
  if (s < NQn) {
    size_t src = ((size_t)(b * NQn + s)) * (3 * Dn) + h * HDn + d;
    q = qkv[src] * scale;
    k = qkv[src + Dn];
    v = qkv[src + 2 * Dn];
  }
  size_t rowmaj = ((size_t)bh * SPn + s) * HDn + d;
  Qf[rowmaj] = (_Float16)q;
  Kf[rowmaj] = (_Float16)k;
  Vt[((size_t)bh * HDn + d) * SPn + s] = (_Float16)v;
}

// ---------------- fused self-attention: one wave per (b,h,16-query tile) ----------------
__global__ void attn_kernel(const _Float16* __restrict__ Qf, const _Float16* __restrict__ Kf,
                            const _Float16* __restrict__ Vt, float* __restrict__ ctx) {
  __shared__ float S[16 * SPn];   // 59392 bytes
  int lane = threadIdx.x;
  int id = blockIdx.x;
  int qt = id % QTn;
  int bh = id / QTn;
  const _Float16* Qb = Qf + (size_t)bh * SPn * HDn;
  const _Float16* Kb = Kf + (size_t)bh * SPn * HDn;
  const _Float16* Vb = Vt + (size_t)bh * HDn * SPn;
  int li = lane & 15, hi = lane >> 4;

  // Q fragment (A layout): row = li, K contiguous (k = hi*16 + i)
  v16h qa;
  int qrow = qt * 16 + li;   // < 912 < SPn, padded rows are zero
  for (int i = 0; i < 16; i += 2) {
    int v = i >> 1;
    int kk = hi * 16 + i;
    qa[i]     = Qb[(size_t)qrow * HDn + kk];
    qa[i + 1] = Qb[(size_t)qrow * HDn + kk + 1];
  }

  // S = Q K^T  (58 tiles of 16 key columns)
  for (int kt = 0; kt < SPn / 16; ++kt) {
    int kcol = kt * 16 + li;
    v16h kb = *(const v16h*)(Kb + (size_t)kcol * HDn + hi * 16);
    v8f c = {};
    c = WMMA16(qa, kb, c);
    for (int r = 0; r < 8; ++r)
      S[(r + hi * 8) * SPn + kt * 16 + li] = c[r];
  }
  __syncthreads();

  // softmax stats for row=li; lane halves split the columns, combine via shfl_xor(16)
  int row = li;
  int c0 = hi * 464;
  int c1 = c0 + 464; if (c1 > NQn) c1 = NQn;
  float mx = -1e30f;
  for (int c = c0; c < c1; ++c) mx = fmaxf(mx, S[row * SPn + c]);
  mx = fmaxf(mx, __shfl_xor(mx, 16, 32));
  float sum = 0.f;
  for (int c = c0; c < c1; ++c) sum += __expf(S[row * SPn + c] - mx);
  sum += __shfl_xor(sum, 16, 32);
  float inv = 1.0f / sum;

  // O = P V : build P fragments on the fly (exp applied per element), V^T gives contiguous B frags
  v8f a0 = {}, a1 = {};
  for (int kt = 0; kt < SPn / 32; ++kt) {
    v16h pa;
    for (int i = 0; i < 16; ++i) {
      int v = i >> 1;
      int col = kt * 32 + hi * 16 + i;
      float s = S[row * SPn + col];
      float p = (col < NQn) ? __expf(s - mx) * inv : 0.f;
      pa[i] = (_Float16)p;
    }
    v16h b0 = *(const v16h*)(Vb + (size_t)li * SPn + kt * 32 + hi * 16);
    v16h b1 = *(const v16h*)(Vb + (size_t)(16 + li) * SPn + kt * 32 + hi * 16);
    a0 = WMMA16(pa, b0, a0);
    a1 = WMMA16(pa, b1, a1);
  }

  int b = bh >> 3, h = bh & 7;
  for (int r = 0; r < 8; ++r) {
    int q = qt * 16 + r + hi * 8;
    if (q < NQn) {
      size_t m = (size_t)(b * NQn + q);
      ctx[m * Dn + h * HDn + li]      = a0[r];
      ctx[m * Dn + h * HDn + 16 + li] = a1[r];
    }
  }
}

// ---------------- deformable bilinear sampler, fused P=4 softmax ----------------
__global__ void sampler_kernel(const float* __restrict__ memory, const float* __restrict__ refp,
                               const float* __restrict__ offs, const float* __restrict__ wlog,
                               float* __restrict__ fused) {
  int idx = blockIdx.x * blockDim.x + threadIdx.x;
  int total = Mn * Dn;
  if (idx >= total) return;
  int hd = idx & 31;
  int h  = (idx >> 5) & 7;
  int m  = idx >> 8;
  int b  = m / NQn;
  int ch = h * HDn + hd;
  float rx = refp[(size_t)m * 2 + 0];
  float ry = refp[(size_t)m * 2 + 1];
  float lg[Pn];
  float lmax = -1e30f;
  for (int p = 0; p < Pn; ++p) { lg[p] = wlog[(size_t)m * (Hn * Pn) + h * Pn + p]; lmax = fmaxf(lmax, lg[p]); }
  float lsum = 0.f;
  for (int p = 0; p < Pn; ++p) { lg[p] = __expf(lg[p] - lmax); lsum += lg[p]; }
  float linv = 1.0f / lsum;
  float acc = 0.f;
  for (int p = 0; p < Pn; ++p) {
    float ox = offs[(size_t)m * (Hn * Pn * 2) + (h * Pn + p) * 2 + 0];
    float oy = offs[(size_t)m * (Hn * Pn * 2) + (h * Pn + p) * 2 + 1];
    float gx = (rx + ox) * (float)BEVn - 0.5f;
    float gy = (ry + oy) * (float)BEVn - 0.5f;
    float x0f = floorf(gx), y0f = floorf(gy);
    int x0 = (int)x0f, y0 = (int)y0f;
    float wx = gx - x0f, wy = gy - y0f;
    float bil = 0.f;
    for (int dy = 0; dy < 2; ++dy) {
      for (int dx = 0; dx < 2; ++dx) {
        int xi = x0 + dx, yi = y0 + dy;
        float w = (dx ? wx : 1.f - wx) * (dy ? wy : 1.f - wy);
        if (xi >= 0 && xi < BEVn && yi >= 0 && yi < BEVn) {
          bil += w * memory[((size_t)b * (BEVn * BEVn) + yi * BEVn + xi) * Dn + ch];
        }
      }
    }
    acc += lg[p] * linv * bil;
  }
  fused[(size_t)m * Dn + ch] = acc;
}

// ---------------- host orchestration ----------------
static inline int ceil_div(int a, int b) { return (a + b - 1) / b; }

extern "C" void kernel_launch(void* const* d_in, const int* in_sizes, int n_in,
                              void* d_out, int out_size, void* d_ws, size_t ws_size,
                              hipStream_t stream) {
  (void)in_sizes; (void)n_in; (void)out_size; (void)ws_size;
  const float* query   = (const float*)d_in[0];
  const float* memory  = (const float*)d_in[1];
  const float* refp    = (const float*)d_in[2];
  const float* qpos    = (const float*)d_in[3];
  const float* in_w    = (const float*)d_in[4];
  const float* in_b    = (const float*)d_in[5];
  const float* out_w   = (const float*)d_in[6];
  const float* out_b   = (const float*)d_in[7];
  const float* off_w   = (const float*)d_in[8];
  const float* off_b   = (const float*)d_in[9];
  const float* wt_w    = (const float*)d_in[10];
  const float* wt_b    = (const float*)d_in[11];
  const float* co_w    = (const float*)d_in[12];
  const float* co_b    = (const float*)d_in[13];
  const float* f_w1    = (const float*)d_in[14];
  const float* f_b1    = (const float*)d_in[15];
  const float* f_w2    = (const float*)d_in[16];
  const float* f_b2    = (const float*)d_in[17];
  const float* n1s     = (const float*)d_in[18];
  const float* n1b     = (const float*)d_in[19];
  const float* n2s     = (const float*)d_in[20];
  const float* n2b     = (const float*)d_in[21];
  const float* n3s     = (const float*)d_in[22];
  const float* n3b     = (const float*)d_in[23];
  float* outp = (float*)d_out;

  char* ws = (char*)d_ws;
  const size_t OFF_APACK = 0;                             // 7200*512*2 = 7,372,800 (max K=512)
  const size_t OFF_BPACK = 7372800;                       // 768*256*2  =   393,216
  const size_t OFF_LN    = 7766016;                       // 7,372,800 (qin / q3)
  const size_t OFF_QKV   = 15138816;                      // 22,118,400 (reused: fused, hidden)
  const size_t OFF_QF    = 37257216;                      // 3,801,088
  const size_t OFF_KF    = 41058304;
  const size_t OFF_VT    = 44859392;
  const size_t OFF_CTX   = 48660480;                      // 7,372,800
  const size_t OFF_Q1    = 56033280;                      // 7,372,800
  const size_t OFF_Q2    = 63406080;                      // 7,372,800
  const size_t OFF_OFFS  = 70778880;                      // 1,843,200
  const size_t OFF_WLOG  = 72622080;                      //   921,600

  _Float16* Apack = (_Float16*)(ws + OFF_APACK);
  _Float16* Bpack = (_Float16*)(ws + OFF_BPACK);
  float* lnbuf  = (float*)(ws + OFF_LN);
  float* qkv    = (float*)(ws + OFF_QKV);
  float* fusedb = (float*)(ws + OFF_QKV);                 // reuse (qkv dead after split)
  float* hidden = (float*)(ws + OFF_QKV + 7372800);       // reuse tail of qkv region
  _Float16* Qf  = (_Float16*)(ws + OFF_QF);
  _Float16* Kf  = (_Float16*)(ws + OFF_KF);
  _Float16* Vt  = (_Float16*)(ws + OFF_VT);
  float* ctx    = (float*)(ws + OFF_CTX);
  float* q1     = (float*)(ws + OFF_Q1);
  float* q2     = (float*)(ws + OFF_Q2);
  float* offsb  = (float*)(ws + OFF_OFFS);
  float* wlog   = (float*)(ws + OFF_WLOG);

  const int TPB = 256;
  auto packA = [&](const float* src, int M, int K) {
    int total = M * K;
    pack_a_kernel<<<ceil_div(total, TPB), TPB, 0, stream>>>(src, Apack, M, K, total);
  };
  auto packB = [&](const float* w, int N, int K) {
    int total = N * K;
    pack_b_kernel<<<ceil_div(total, TPB), TPB, 0, stream>>>(w, Bpack, N, K, total);
  };
  auto gemm = [&](const float* bias, const float* res, float* out, int M, int N, int K, int act) {
    if ((N & 63) == 0) {
      dim3 g(N / 64, M / 16);
      gemm_wmma4_kernel<<<g, 128, 0, stream>>>(Apack, Bpack, bias, res, out, M, N, K, act);
    } else {
      dim3 g(N / 16, M / 16);
      gemm_wmma_kernel<<<g, 32, 0, stream>>>(Apack, Bpack, bias, res, out, M, N, K, act);
    }
  };

  // 1) q_in = LN(query + query_pos; n1)
  ln_kernel<<<Mn, 256, 0, stream>>>(query, qpos, n1s, n1b, lnbuf);
  // 2-4) qkv = q_in @ in_w^T + in_b
  packA(lnbuf, Mn, Dn);
  packB(in_w, 3 * Dn, Dn);
  gemm(in_b, nullptr, qkv, Mn, 3 * Dn, Dn, 0);
  // 5) split into Q (scaled) / K / V^T, padded to 928
  {
    int total = Bn * Hn * SPn * HDn;
    qkv_split_kernel<<<ceil_div(total, TPB), TPB, 0, stream>>>(qkv, Qf, Kf, Vt);
  }
  // 6) fused self-attention
  attn_kernel<<<Bn * Hn * QTn, 32, 0, stream>>>(Qf, Kf, Vt, ctx);
  // 7-9) q1 = query + ctx @ out_w^T + out_b
  packA(ctx, Mn, Dn);
  packB(out_w, Dn, Dn);
  gemm(out_b, query, q1, Mn, Dn, Dn, 0);
  // 10) q2 = LN(q1 + query_pos; n2)
  ln_kernel<<<Mn, 256, 0, stream>>>(q1, qpos, n2s, n2b, q2);
  // 11-13) offsets = tanh(q2 @ off_w^T + off_b) * 0.2
  packA(q2, Mn, Dn);
  packB(off_w, Hn * Pn * 2, Dn);
  gemm(off_b, nullptr, offsb, Mn, Hn * Pn * 2, Dn, 2);
  // 14-15) weight logits = q2 @ wt_w^T + wt_b   (N=32 -> single-wave kernel)
  packB(wt_w, Hn * Pn, Dn);
  gemm(wt_b, nullptr, wlog, Mn, Hn * Pn, Dn, 0);
  // 16) deformable bilinear sampling -> fused
  {
    int total = Mn * Dn;
    sampler_kernel<<<ceil_div(total, TPB), TPB, 0, stream>>>(memory, refp, offsb, wlog, fusedb);
  }
  // 17-18) query2 = q1 + fused @ co_w^T + co_b
  packA(fusedb, Mn, Dn);
  packB(co_w, Dn, Dn);
  gemm(co_b, q1, q2, Mn, Dn, Dn, 0);
  // 19) q3 = LN(q2; n3)
  ln_kernel<<<Mn, 256, 0, stream>>>(q2, nullptr, n3s, n3b, lnbuf);
  // 20-21) hidden = relu(q3 @ f_w1^T + f_b1)
  packA(lnbuf, Mn, Dn);
  packB(f_w1, FFNn, Dn);
  gemm(f_b1, nullptr, hidden, Mn, FFNn, Dn, 1);
  // 22-23) out = q2 + hidden @ f_w2^T + f_b2
  packA(hidden, Mn, FFNn);
  packB(f_w2, Dn, FFNn);
  gemm(f_b2, q2, outp, Mn, Dn, FFNn, 0);
}
